// HAT_9122510537341
// MI455X (gfx1250) — compile-verified, matrix operand order checked by reading, not run on hardware
//
#include <hip/hip_runtime.h>
#include <math.h>

// ---------------------------------------------------------------------------
// HAT block for MI455X (gfx1250, wave32).  All dense algebra runs through
// v_wmma_f32_16x16x32_f16 (f16 in / f32 accumulate).  Attention is fused per
// (item, head, 64-row tile) with scores+softmax kept in LDS.
// Roofline: ~210 GFLOP of GEMM vs ~1 GB HBM traffic @ 23.3 TB/s -> matrix-
// core bound, so stage f32 -> f16 once into LDS and ride the WMMA pipe.
// ---------------------------------------------------------------------------

typedef __attribute__((ext_vector_type(16))) _Float16 v16h;
typedef __attribute__((ext_vector_type(8)))  _Float16 v8h;
typedef __attribute__((ext_vector_type(4)))  _Float16 v4h;
typedef __attribute__((ext_vector_type(8)))  float    v8f;
typedef __attribute__((ext_vector_type(4)))  float    v4f;

namespace hat {
constexpr int B     = 32;
constexpr int SR    = 8;
constexpr int CS    = 2;
constexpr int WS    = 7;
constexpr int D     = 256;
constexpr int NH    = 8;
constexpr int HD    = 32;
constexpr int NW    = SR * SR;               // 64
constexpr int NWIN  = B * NW;                // 2048
constexpr int NT_W  = CS * CS + WS * WS;     // 53 tokens / window
constexpr int NT_H  = (SR * CS) * (SR * CS); // 256 hier tokens
constexpr int MW    = NWIN * NT_W;           // 108544 window-stage rows (64 | MW)
constexpr int MH    = B * NT_H;              // 8192 hier-stage rows     (64 | MH)
constexpr int MLP_H = 4 * D;                 // 1024
constexpr float SCALE = 0.17677669529663687f; // 32^-0.5
}

// ---------------------------------------------------------------------------
// WMMA helpers — wave32 16-bit operand layout (CDNA5 ISA 7.12.2):
// A 16x32 f16: lane L holds row M=L&15; halves 0..7 -> K = base+0..7,
// halves 8..15 -> K = base+16..23, base = (L>=16) ? 8 : 0.  B uses the same
// pattern with N in place of M, so B is staged as [N][K] in LDS.  Both
// 8-half segments are contiguous -> two ds_load_b128 per fragment.
// ---------------------------------------------------------------------------
__device__ inline v16h load_frag(const _Float16* base, int ld, int row0, int k0, int lane) {
  int r  = row0 + (lane & 15);
  int kb = k0 + ((lane >> 4) << 3);
  const _Float16* p = base + r * ld + kb;
  v8h lo = *(const v8h*)(p);
  v8h hi = *(const v8h*)(p + 16);
  v16h f;
#pragma unroll
  for (int i = 0; i < 8; ++i) { f[i] = lo[i]; f[i + 8] = hi[i]; }
  return f;
}

__device__ inline v8f wmma16(v16h a, v16h b, v8f c) {
  return __builtin_amdgcn_wmma_f32_16x16x32_f16(false, a, false, b, (short)0, c, false, false);
}

__device__ inline v8f vzero8() {
  v8f z = {0.f, 0.f, 0.f, 0.f, 0.f, 0.f, 0.f, 0.f};
  return z;
}

__device__ inline v4h cvt4(v4f v) {
  v4h h;
  h[0] = (_Float16)v[0]; h[1] = (_Float16)v[1];
  h[2] = (_Float16)v[2]; h[3] = (_Float16)v[3];
  return h;
}

// ---------------------------------------------------------------------------
// Small MLP (2 -> 512 -> out_dim), one block per row.
// mode 0: pos1d coords on p x p grid.  mode 1: log-CPB table coords, ws = p.
// ---------------------------------------------------------------------------
__global__ void small_mlp_kernel(const float* __restrict__ w1, const float* __restrict__ b1,
                                 const float* __restrict__ w2, float* __restrict__ out,
                                 int out_dim, int mode, int p) {
  __shared__ float hid[512];
  int row = blockIdx.x, tid = threadIdx.x;
  float v0, v1;
  if (mode == 0) {
    int cc = p >> 1;
    v0 = (float)(row / p - cc) / (float)cc;
    v1 = (float)(row % p - cc) / (float)cc;
  } else {
    int n = 2 * p - 1;
    float th = (float)(row / n) - (float)(p - 1);
    float tw = (float)(row % n) - (float)(p - 1);
    float z0 = th / (float)(p - 1) * 8.0f;
    float z1 = tw / (float)(p - 1) * 8.0f;
    v0 = copysignf(log2f(fabsf(z0) + 1.0f) * (1.0f / 3.0f), z0);
    v1 = copysignf(log2f(fabsf(z1) + 1.0f) * (1.0f / 3.0f), z1);
  }
  for (int h = tid; h < 512; h += blockDim.x)
    hid[h] = fmaxf(v0 * w1[h] + v1 * w1[512 + h] + b1[h], 0.0f);
  __syncthreads();
  for (int o = tid; o < out_dim; o += blockDim.x) {
    float acc = 0.f;
    for (int h = 0; h < 512; ++h) acc += hid[h] * w2[h * out_dim + o];
    out[row * out_dim + o] = acc;
  }
}

// Expand bias table (R x NH) -> (NH, nt, nt) with left/top zero padding.
__global__ void bias_expand_kernel(const float* __restrict__ bt, float* __restrict__ bias,
                                   int ws, int pad, int nt) {
  int idx = blockIdx.x * blockDim.x + threadIdx.x;
  int total = hat::NH * nt * nt;
  if (idx >= total) return;
  int h = idx / (nt * nt);
  int rem = idx % (nt * nt);
  int i = rem / nt, j = rem % nt;
  float v = 0.f;
  if (i >= pad && j >= pad) {
    int p = i - pad, q = j - pad;
    int ph = p / ws, pw = p % ws, qh = q / ws, qw = q % ws;
    int id = (ph - qh + ws - 1) * (2 * ws - 1) + (pw - qw + ws - 1);
    float b = bt[id * hat::NH + h];
    v = 16.0f / (1.0f + __expf(-b));
  }
  bias[idx] = v;
}

// ct (B, 256, D) -> pooled (B, 64, D): mean over 4 consecutive CT tokens.
__global__ void pool_kernel(const float* __restrict__ ct, float* __restrict__ pooled) {
  int i4 = blockIdx.x * blockDim.x + threadIdx.x;  // (b,g,d/4): 32*64*64
  int d4 = (i4 & 63) << 2, g = (i4 >> 6) & 63, b = i4 >> 12;
  const float* base = ct + (((b << 8) + (g << 2)) << 8) + d4;
  v4f s = (*(const v4f*)(base) + *(const v4f*)(base + 256) +
           *(const v4f*)(base + 512) + *(const v4f*)(base + 768)) * 0.25f;
  *(v4f*)(pooled + ((long)i4 << 2)) = s;
}

// pooled (B, 8x8, D) -> bilinear (half-pixel) 16x16, +hpe -> c (B, 256, D)
__global__ void resize_pe_kernel(const float* __restrict__ pooled, const float* __restrict__ hpe,
                                 float* __restrict__ c) {
  int i4 = blockIdx.x * blockDim.x + threadIdx.x;  // (b,y,x,d/4): 32*16*16*64
  int d4 = (i4 & 63) << 2, x = (i4 >> 6) & 15, y = (i4 >> 10) & 15, b = i4 >> 14;
  float fy = y * 0.5f - 0.25f, fx = x * 0.5f - 0.25f;
  int y0 = (int)floorf(fy); float ty = fy - (float)y0;
  int x0 = (int)floorf(fx); float tx = fx - (float)x0;
  int y0c = min(max(y0, 0), 7), y1c = min(max(y0 + 1, 0), 7);
  int x0c = min(max(x0, 0), 7), x1c = min(max(x0 + 1, 0), 7);
  const float* pb = pooled + (b << 14) + d4;
  v4f p00 = *(const v4f*)(pb + ((y0c * 8 + x0c) << 8));
  v4f p01 = *(const v4f*)(pb + ((y0c * 8 + x1c) << 8));
  v4f p10 = *(const v4f*)(pb + ((y1c * 8 + x0c) << 8));
  v4f p11 = *(const v4f*)(pb + ((y1c * 8 + x1c) << 8));
  v4f v = (p00 * (1.f - tx) + p01 * tx) * (1.f - ty) + (p10 * (1.f - tx) + p11 * tx) * ty;
  v4f pe = *(const v4f*)(hpe + ((y << 4) + x) * 256 + d4);
  *(v4f*)(c + (((b << 8) + (y << 4) + x) << 8) + d4) = v + pe;
}

// LayerNorm over D=256, one block (256 threads) per row.
__global__ void ln_kernel(const float* __restrict__ in, const float* __restrict__ g,
                          const float* __restrict__ b, float* __restrict__ out) {
  __shared__ float red[256];
  int row = blockIdx.x, tid = threadIdx.x;
  long base = (long)row * 256;
  float v = in[base + tid];
  red[tid] = v;
  __syncthreads();
  for (int s = 128; s > 0; s >>= 1) { if (tid < s) red[tid] += red[tid + s]; __syncthreads(); }
  float mean = red[0] * (1.0f / 256.0f);
  __syncthreads();
  float dv = v - mean;
  red[tid] = dv * dv;
  __syncthreads();
  for (int s = 128; s > 0; s >>= 1) { if (tid < s) red[tid] += red[tid + s]; __syncthreads(); }
  float var = red[0] * (1.0f / 256.0f);
  out[base + tid] = dv * rsqrtf(var + 1e-5f) * g[tid] + b[tid];
}

// ---------------------------------------------------------------------------
// Generic WMMA GEMM: out = epilogue(A(MxK,f32) @ W(KxN,f32)).
// 64x64 tile / block (128 threads = 4 waves, each wave 2x2 16x16 tiles),
// BK = 32.  REQUIRES: 64 | M, 64 | N, 32 | K (true for every launch here),
// so staging and the epilogue are branch-free and fully vectorized.
// act: 0 = none, 1 = exact GELU.  bias/res optional (res/out may alias).
// ---------------------------------------------------------------------------
__global__ void gemm_wmma_kernel(const float* __restrict__ A, const float* __restrict__ W,
                                 const float* __restrict__ bias, const float* res,
                                 float* out, int M, int N, int K, int act) {
  __shared__ _Float16 Ah[64 * 32];   // [m][k]
  __shared__ _Float16 Bh[64 * 32];   // [n][k] (transposed stage)
  int tid = threadIdx.x, lane = tid & 31, wv = tid >> 5;
  int wm = wv >> 1, wn = wv & 1;
  int gm0 = blockIdx.y * 64, gn0 = blockIdx.x * 64;
  v8f acc[2][2];
  acc[0][0] = vzero8(); acc[0][1] = vzero8();
  acc[1][0] = vzero8(); acc[1][1] = vzero8();
  for (int kk = 0; kk < K; kk += 32) {
    if (kk + 32 < K) {  // speculative prefetch of next K tile -> global_prefetch_b8
      __builtin_prefetch(&A[(long)(gm0 + (tid >> 1)) * K + kk + 32], 0, 3);
      __builtin_prefetch(&W[(long)(kk + 32 + (tid & 31)) * N + gn0], 0, 3);
    }
    // stage A: 512 float4 loads -> packed v4h -> ds_store_b64 (branch-free)
#pragma unroll
    for (int i = 0; i < 4; ++i) {
      int e4 = tid + i * 128;          // 0..511
      int r  = e4 >> 3;                // row 0..63
      int c4 = (e4 & 7) << 2;          // col 0,4,..,28
      v4f v = *(const v4f*)(A + (long)(gm0 + r) * K + kk + c4);
      *(v4h*)(Ah + r * 32 + c4) = cvt4(v);
    }
    // stage B transposed [n][k]: coalesced float4 along n, 4 b16 LDS stores
#pragma unroll
    for (int i = 0; i < 4; ++i) {
      int e4 = tid + i * 128;
      int k  = e4 >> 4;                // 0..31
      int n0 = (e4 & 15) << 2;         // 0..60
      v4f v = *(const v4f*)(W + (long)(kk + k) * N + gn0 + n0);
      Bh[(n0 + 0) * 32 + k] = (_Float16)v[0];
      Bh[(n0 + 1) * 32 + k] = (_Float16)v[1];
      Bh[(n0 + 2) * 32 + k] = (_Float16)v[2];
      Bh[(n0 + 3) * 32 + k] = (_Float16)v[3];
    }
    __syncthreads();
    v16h a0 = load_frag(Ah, 32, wm * 32, 0, lane);
    v16h a1 = load_frag(Ah, 32, wm * 32 + 16, 0, lane);
    v16h b0 = load_frag(Bh, 32, wn * 32, 0, lane);
    v16h b1 = load_frag(Bh, 32, wn * 32 + 16, 0, lane);
    acc[0][0] = wmma16(a0, b0, acc[0][0]);
    acc[0][1] = wmma16(a0, b1, acc[0][1]);
    acc[1][0] = wmma16(a1, b0, acc[1][0]);
    acc[1][1] = wmma16(a1, b1, acc[1][1]);
    __syncthreads();
  }
  // C/D layout: VGPR e -> M = e + (lane>=16 ? 8 : 0), N = lane & 15
#pragma unroll
  for (int i = 0; i < 2; ++i)
#pragma unroll
    for (int j = 0; j < 2; ++j) {
      int rb  = gm0 + wm * 32 + i * 16 + ((lane >> 4) << 3);
      int col = gn0 + wn * 32 + j * 16 + (lane & 15);
      float bv = bias ? bias[col] : 0.f;
#pragma unroll
      for (int e = 0; e < 8; ++e) {
        long r = rb + e;
        float v = acc[i][j][e] + bv;
        if (act == 1) v = 0.5f * v * (1.0f + erff(v * 0.70710678f));
        if (res) v += res[r * N + col];
        out[r * N + col] = v;
      }
    }
}

// ---------------------------------------------------------------------------
// Fused attention for one (item, head, 64-row tile).  NPAD = padded token
// count (64 for windows, 256 for hier).  qkv rows are 768 wide
// [q|k|v] x [head*32+d].  Scores + softmax live in LDS (dynamic).
// ---------------------------------------------------------------------------
template <int NPAD>
__global__ void attn_wmma_kernel(const float* __restrict__ qkv, float* __restrict__ o,
                                 const float* __restrict__ bias, int n_valid,
                                 int tok_stride) {
  extern __shared__ char smem_raw[];
  _Float16* Qh  = (_Float16*)smem_raw;          // [64][32]
  _Float16* Kh  = Qh + 64 * 32;                 // [NPAD][32]
  _Float16* VhT = Kh + NPAD * 32;               // [32][NPAD]  (B-layout for P@V)
  float*    S   = (float*)(VhT + 32 * NPAD);    // [64][NPAD]
  _Float16* P   = (_Float16*)(S + 64 * NPAD);   // [64][NPAD]
  int tid = threadIdx.x, lane = tid & 31, wv = tid >> 5;
  int h = blockIdx.y, m0 = blockIdx.z * 64;
  long tok_base = (long)blockIdx.x * tok_stride;
  v4f z4 = {0.f, 0.f, 0.f, 0.f};

  // Q: 64x32 -> 512 float4, vectorized with per-row validity
#pragma unroll
  for (int i = 0; i < 4; ++i) {
    int e4 = tid + i * 128;
    int r = e4 >> 3, d4 = (e4 & 7) << 2;
    int m = m0 + r;
    v4f v = (m < n_valid) ? *(const v4f*)(qkv + (tok_base + m) * 768 + h * 32 + d4) : z4;
    *(v4h*)(Qh + r * 32 + d4) = cvt4(v);
  }
  // K row-major [n][d]; V transposed [d][n]
#pragma unroll
  for (int i = 0; i < NPAD / 16; ++i) {
    int e4 = tid + i * 128;
    int n = e4 >> 3, d4 = (e4 & 7) << 2;
    bool ok = (n < n_valid);
    v4f kv = ok ? *(const v4f*)(qkv + (tok_base + n) * 768 + 256 + h * 32 + d4) : z4;
    v4f vv = ok ? *(const v4f*)(qkv + (tok_base + n) * 768 + 512 + h * 32 + d4) : z4;
    *(v4h*)(Kh + n * 32 + d4) = cvt4(kv);
    VhT[(d4 + 0) * NPAD + n] = (_Float16)vv[0];
    VhT[(d4 + 1) * NPAD + n] = (_Float16)vv[1];
    VhT[(d4 + 2) * NPAD + n] = (_Float16)vv[2];
    VhT[(d4 + 3) * NPAD + n] = (_Float16)vv[3];
  }
  __syncthreads();

  // S = Q @ K^T * scale + bias (Kh already [n][d] == B-layout [N][K])
  v16h aq = load_frag(Qh, 32, wv * 16, 0, lane);
#pragma unroll
  for (int nt = 0; nt < NPAD / 16; ++nt) {
    v16h bk = load_frag(Kh, 32, nt * 16, 0, lane);
    v8f acc = wmma16(aq, bk, vzero8());
    int rl0 = wv * 16 + ((lane >> 4) << 3);
    int cn  = nt * 16 + (lane & 15);
#pragma unroll
    for (int e = 0; e < 8; ++e) {
      int rl = rl0 + e, gm = m0 + rl;
      float s;
      if (cn < n_valid) {
        s = acc[e] * hat::SCALE;
        if (gm < n_valid) s += bias[((long)h * n_valid + gm) * n_valid + cn];
      } else {
        s = -1e30f;
      }
      S[rl * NPAD + cn] = s;
    }
  }
  __syncthreads();

  if (tid < 64) {  // row softmax (exp stored back into S, single pass)
    float mx = -1e30f;
    for (int cx = 0; cx < NPAD; ++cx) mx = fmaxf(mx, S[tid * NPAD + cx]);
    float sum = 0.f;
    for (int cx = 0; cx < NPAD; ++cx) {
      float e = __expf(S[tid * NPAD + cx] - mx);
      S[tid * NPAD + cx] = e;
      sum += e;
    }
    float inv = 1.0f / sum;
    for (int cx = 0; cx < NPAD; ++cx)
      P[tid * NPAD + cx] = (_Float16)(S[tid * NPAD + cx] * inv);
  }
  __syncthreads();

  // O = P @ V   (VhT is [d][n] == B-layout [N][K] with K = token)
#pragma unroll
  for (int nt = 0; nt < 2; ++nt) {
    v8f acc = vzero8();
#pragma unroll
    for (int ks = 0; ks < NPAD / 32; ++ks) {
      v16h ap = load_frag(P, NPAD, wv * 16, ks * 32, lane);
      v16h bv = load_frag(VhT, NPAD, nt * 16, ks * 32, lane);
      acc = wmma16(ap, bv, acc);
    }
    int rl0 = wv * 16 + ((lane >> 4) << 3);
    int cd  = nt * 16 + (lane & 15);
#pragma unroll
    for (int e = 0; e < 8; ++e) {
      int gm = m0 + rl0 + e;
      if (gm < n_valid) o[(tok_base + gm) * 256 + h * 32 + cd] = acc[e];
    }
  }
}

// Build xcat (2048, 53, 256): tokens 0..3 gathered from hier state c,
// tokens 4..52 = x + pe49.  float4 over channels.
__global__ void build_xcat_kernel(const float* __restrict__ x, const float* __restrict__ pe,
                                  const float* __restrict__ c, float* __restrict__ xcat) {
  long i4 = (long)blockIdx.x * blockDim.x + threadIdx.x;   // MW * 64
  int d4 = ((int)i4 & 63) << 2;
  int t = (int)(i4 >> 6);
  int l = t % hat::NT_W;
  int w = t / hat::NT_W;
  v4f v;
  if (l < 4) {
    int b = w >> 6, wi = w & 63, s1 = wi >> 3, s2 = wi & 7;
    int cy = l >> 1, cx = l & 1;
    int ctok = (s1 * 2 + cy) * 16 + (s2 * 2 + cx);
    v = *(const v4f*)(c + ((long)(b * 256 + ctok) << 8) + d4);
  } else {
    v = *(const v4f*)(x + ((long)w * 49 + (l - 4)) * 256 + d4) +
        *(const v4f*)(pe + (l - 4) * 256 + d4);
  }
  *(v4f*)(xcat + ((long)t << 8) + d4) = v;
}

// Split xcat -> d_out = [xo (2048,49,256) | ctr (2048*4,256)]
__global__ void copy_out_kernel(const float* __restrict__ xcat, float* __restrict__ out) {
  long i4 = (long)blockIdx.x * blockDim.x + threadIdx.x;   // MW * 64
  int d4 = ((int)i4 & 63) << 2;
  int t = (int)(i4 >> 6);
  int l = t % hat::NT_W;
  int w = t / hat::NT_W;
  v4f v = *(const v4f*)(xcat + ((long)t << 8) + d4);
  if (l >= 4) *(v4f*)(out + ((long)w * 49 + (l - 4)) * 256 + d4) = v;
  else        *(v4f*)(out + (long)hat::NWIN * 49 * 256 + ((long)w * 4 + l) * 256 + d4) = v;
}

// ---------------------------------------------------------------------------
extern "C" void kernel_launch(void* const* d_in, const int* in_sizes, int n_in,
                              void* d_out, int out_size, void* d_ws, size_t ws_size,
                              hipStream_t stream) {
  using namespace hat;
  (void)in_sizes; (void)n_in; (void)out_size; (void)ws_size;
  const float* x      = (const float*)d_in[0];
  const float* ct     = (const float*)d_in[1];
  const float* pe_w1  = (const float*)d_in[2];
  const float* pe_b1  = (const float*)d_in[3];
  const float* pe_w2  = (const float*)d_in[4];
  const float* hpe_w1 = (const float*)d_in[5];
  const float* hpe_b1 = (const float*)d_in[6];
  const float* hpe_w2 = (const float*)d_in[7];
  const float* n1_g   = (const float*)d_in[8];
  const float* n1_b   = (const float*)d_in[9];
  const float* n2_g   = (const float*)d_in[10];
  const float* n2_b   = (const float*)d_in[11];
  const float* hn1_g  = (const float*)d_in[12];
  const float* hn1_b  = (const float*)d_in[13];
  const float* hn2_g  = (const float*)d_in[14];
  const float* hn2_b  = (const float*)d_in[15];
  const float* qkv_w  = (const float*)d_in[16];
  const float* proj_w = (const float*)d_in[17];
  const float* proj_b = (const float*)d_in[18];
  const float* cpb_w1 = (const float*)d_in[19];
  const float* cpb_b1 = (const float*)d_in[20];
  const float* cpb_w2 = (const float*)d_in[21];
  const float* hqkv_w = (const float*)d_in[22];
  const float* hproj_w= (const float*)d_in[23];
  const float* hproj_b= (const float*)d_in[24];
  const float* hcpb_w1= (const float*)d_in[25];
  const float* hcpb_b1= (const float*)d_in[26];
  const float* hcpb_w2= (const float*)d_in[27];
  const float* fc1_w  = (const float*)d_in[28];
  const float* fc1_b  = (const float*)d_in[29];
  const float* fc2_w  = (const float*)d_in[30];
  const float* fc2_b  = (const float*)d_in[31];
  const float* hfc1_w = (const float*)d_in[32];
  const float* hfc1_b = (const float*)d_in[33];
  const float* hfc2_w = (const float*)d_in[34];
  const float* hfc2_b = (const float*)d_in[35];
  float* out = (float*)d_out;

  // Workspace carving (~281M floats = ~1.13 GB; MI455X has 432 GiB HBM).
  float* ws = (float*)d_ws;
  size_t off = 0;
  auto alloc = [&](size_t n) { float* p = ws + off; off += (n + 63) & ~(size_t)63; return p; };
  float* pe49   = alloc(49 * 256);
  float* hpe256 = alloc(256 * 256);
  float* btH    = alloc(961 * NH);
  float* btW    = alloc(169 * NH);
  float* hbias  = alloc((size_t)NH * NT_H * NT_H);
  float* wbias  = alloc((size_t)NH * NT_W * NT_W);
  float* pooled = alloc((size_t)B * 64 * 256);
  float* c      = alloc((size_t)MH * 256);
  float* xcat   = alloc((size_t)MW * 256);
  float* lnb    = alloc((size_t)MW * 256);   // reused by both stages
  float* qkvb   = alloc((size_t)MW * 768);
  float* ob     = alloc((size_t)MW * 256);
  float* mlpb   = alloc((size_t)MW * 1024);

  // -- small precompute ----------------------------------------------------
  small_mlp_kernel<<<49, 256, 0, stream>>>(pe_w1, pe_b1, pe_w2, pe49, 256, 0, 7);
  small_mlp_kernel<<<256, 256, 0, stream>>>(hpe_w1, hpe_b1, hpe_w2, hpe256, 256, 0, 16);
  small_mlp_kernel<<<961, 256, 0, stream>>>(hcpb_w1, hcpb_b1, hcpb_w2, btH, 8, 1, 16);
  small_mlp_kernel<<<169, 256, 0, stream>>>(cpb_w1, cpb_b1, cpb_w2, btW, 8, 1, 7);
  bias_expand_kernel<<<(NH * NT_H * NT_H + 255) / 256, 256, 0, stream>>>(btH, hbias, 16, 0, NT_H);
  bias_expand_kernel<<<(NH * NT_W * NT_W + 255) / 256, 256, 0, stream>>>(btW, wbias, 7, 4, NT_W);
  pool_kernel<<<(B * 64 * 64) / 256, 256, 0, stream>>>(ct, pooled);
  resize_pe_kernel<<<(B * 256 * 64) / 256, 256, 0, stream>>>(pooled, hpe256, c);

  // attention dynamic-LDS sizes: 4096 + 512 * NPAD bytes
  const size_t shmH = 4096 + 512 * 256;  // 132 KB (<= 320 KB/WGP; 2 WGs/WGP)
  const size_t shmW = 4096 + 512 * 64;   //  36 KB
  (void)hipFuncSetAttribute((const void*)attn_wmma_kernel<256>,
                            hipFuncAttributeMaxDynamicSharedMemorySize, (int)shmH);

  // -- hierarchical (carrier token) block: 8192 rows ----------------------
  ln_kernel<<<MH, 256, 0, stream>>>(c, hn1_g, hn1_b, lnb);
  gemm_wmma_kernel<<<dim3(768 / 64, MH / 64), 128, 0, stream>>>(
      lnb, hqkv_w, nullptr, nullptr, qkvb, MH, 768, 256, 0);
  attn_wmma_kernel<256><<<dim3(B, NH, NT_H / 64), 128, shmH, stream>>>(
      qkvb, ob, hbias, NT_H, NT_H);
  gemm_wmma_kernel<<<dim3(256 / 64, MH / 64), 128, 0, stream>>>(
      ob, hproj_w, hproj_b, c, c, MH, 256, 256, 0);
  ln_kernel<<<MH, 256, 0, stream>>>(c, hn2_g, hn2_b, lnb);
  gemm_wmma_kernel<<<dim3(MLP_H / 64, MH / 64), 128, 0, stream>>>(
      lnb, hfc1_w, hfc1_b, nullptr, mlpb, MH, MLP_H, 256, 1);
  gemm_wmma_kernel<<<dim3(256 / 64, MH / 64), 128, 0, stream>>>(
      mlpb, hfc2_w, hfc2_b, c, c, MH, 256, MLP_H, 0);

  // -- window block: 2048 windows x 53 tokens = 108544 rows ----------------
  build_xcat_kernel<<<(MW * 64) / 256, 256, 0, stream>>>(x, pe49, c, xcat);
  ln_kernel<<<MW, 256, 0, stream>>>(xcat, n1_g, n1_b, lnb);
  gemm_wmma_kernel<<<dim3(768 / 64, MW / 64), 128, 0, stream>>>(
      lnb, qkv_w, nullptr, nullptr, qkvb, MW, 768, 256, 0);
  attn_wmma_kernel<64><<<dim3(NWIN, NH, 1), 128, shmW, stream>>>(
      qkvb, ob, wbias, NT_W, NT_W);
  gemm_wmma_kernel<<<dim3(256 / 64, MW / 64), 128, 0, stream>>>(
      ob, proj_w, proj_b, xcat, xcat, MW, 256, 256, 0);
  ln_kernel<<<MW, 256, 0, stream>>>(xcat, n2_g, n2_b, lnb);
  gemm_wmma_kernel<<<dim3(MLP_H / 64, MW / 64), 128, 0, stream>>>(
      lnb, fc1_w, fc1_b, nullptr, mlpb, MW, MLP_H, 256, 1);
  gemm_wmma_kernel<<<dim3(256 / 64, MW / 64), 128, 0, stream>>>(
      mlpb, fc2_w, fc2_b, xcat, xcat, MW, 256, MLP_H, 0);

  // -- emit (xo, ctr) ------------------------------------------------------
  copy_out_kernel<<<(MW * 64) / 256, 256, 0, stream>>>(xcat, out);
}
